// Attention_26225070309829
// MI455X (gfx1250) — matrix operand order repaired and checked by reading, not compiled, not run on hardware
//
#include <hip/hip_runtime.h>
#include <hip/hip_bf16.h>
#include <math.h>

#define EMBED    768
#define HEADS    12
#define HEAD_DIM 64
#define BATCH    16
#define SEQ      1024
#define QKV_N    (3*EMBED)      // 2304
#define MTOT     (BATCH*SEQ)    // 16384

typedef __attribute__((ext_vector_type(16))) __bf16 bf16x16;
typedef __attribute__((ext_vector_type(8)))  __bf16 bf16x8;
typedef __attribute__((ext_vector_type(4)))  __bf16 bf16x4;
typedef __attribute__((ext_vector_type(8)))  float  f32x8;

// ---- async global->LDS staging (CDNA5); fall back to ds_store path --------
#if defined(__has_builtin)
#  if __has_builtin(__builtin_amdgcn_global_load_async_to_lds_b128)
#    define USE_ASYNC 1
#  endif
#endif
#ifndef USE_ASYNC
#  define USE_ASYNC 0
#endif

#if USE_ASYNC
typedef int v4i_ __attribute__((vector_size(16)));
typedef __attribute__((address_space(1))) v4i_ gv4i;   // prints as "int4 __device__*"
typedef __attribute__((address_space(3))) v4i_ lv4i;   // LDS int4
// 16B per-lane copy, tracked by ASYNCcnt. Generic LDS addr[31:0] == LDS offset.
static __device__ inline void copy16(const __bf16* g, __bf16* l) {
  __builtin_amdgcn_global_load_async_to_lds_b128(
      (gv4i*)(uintptr_t)g, (lv4i*)(unsigned int)(uintptr_t)l, 0, 0);
}
static __device__ inline void wait_stage() {
#  if __has_builtin(__builtin_amdgcn_s_wait_asynccnt)
  __builtin_amdgcn_s_wait_asynccnt(0);
#  else
  asm volatile("s_wait_asynccnt 0x0" ::: "memory");
#  endif
}
#else
static __device__ inline void copy16(const __bf16* g, __bf16* l) {
  *(bf16x8*)l = *(const bf16x8*)g;
}
static __device__ inline void wait_stage() {}
#endif

static __device__ inline f32x8 zero8() {
  f32x8 z;
#pragma unroll
  for (int i = 0; i < 8; ++i) z[i] = 0.f;
  return z;
}
static __device__ inline bf16x16 frag_cat(bf16x8 lo, bf16x8 hi) {
  return __builtin_shufflevector(lo, hi, 0,1,2,3,4,5,6,7,8,9,10,11,12,13,14,15);
}
// A-operand (16x32 bf16): lane row = lane%16; elems = K[8h..8h+7], K[16+8h..+7]
static __device__ inline bf16x16 loadA32(const __bf16* p, int lhalf) {
  bf16x8 lo = *(const bf16x8*)(p + 8*lhalf);
  bf16x8 hi = *(const bf16x8*)(p + 16 + 8*lhalf);
  return frag_cat(lo, hi);
}
// B-operand (32x16 bf16): lane col = lane%16; elems = K[8h..8h+7], K[16+8h..+7]
static __device__ inline bf16x16 loadB32(const __bf16* p, int lhalf) {
  bf16x8 lo = *(const bf16x8*)(p + 8*lhalf);
  bf16x8 hi = *(const bf16x8*)(p + 16 + 8*lhalf);
  return frag_cat(lo, hi);
}
static __device__ inline f32x8 wmma_bf16(bf16x16 a, bf16x16 b, f32x8 c) {
  return __builtin_amdgcn_wmma_f32_16x16x32_bf16(false, a, false, b, (short)0, c,
                                                 false, false);
}
static __device__ inline float rmax16(float v) {
  v = fmaxf(v, __shfl_xor(v, 1, 32));
  v = fmaxf(v, __shfl_xor(v, 2, 32));
  v = fmaxf(v, __shfl_xor(v, 4, 32));
  v = fmaxf(v, __shfl_xor(v, 8, 32));
  return v;
}
static __device__ inline float rsum16(float v) {
  v += __shfl_xor(v, 1, 32);
  v += __shfl_xor(v, 2, 32);
  v += __shfl_xor(v, 4, 32);
  v += __shfl_xor(v, 8, 32);
  return v;
}

// ---------------- prepass: fp32 -> bf16 elementwise ------------------------
__global__ __launch_bounds__(256)
void cvt_kernel(const float* __restrict__ in, __bf16* __restrict__ out) {
  const size_t i = ((size_t)blockIdx.x * blockDim.x + threadIdx.x) * 4;
  const float4 v = *(const float4*)(in + i);
  bf16x4 h; h[0] = (__bf16)v.x; h[1] = (__bf16)v.y; h[2] = (__bf16)v.z; h[3] = (__bf16)v.w;
  *(bf16x4*)(out + i) = h;
}

// ---------------- prepass: fp32 [K][N] -> bf16 transposed [N][K] -----------
__global__ __launch_bounds__(256)
void transpose_cvt_kernel(const float* __restrict__ w, __bf16* __restrict__ wt,
                          int K, int N) {
  __shared__ __bf16 ts[32][34];
  const int tid = threadIdx.x;
  const int n0 = blockIdx.x * 32, k0 = blockIdx.y * 32;
  const int j = tid & 31, i0 = tid >> 5;       // j: fast index, i0: 0..7
#pragma unroll
  for (int p = 0; p < 4; ++p) {
    const int i = i0 + p * 8;
    ts[j][i] = (__bf16)w[(size_t)(k0 + i) * N + n0 + j];   // coalesced over n
  }
  __syncthreads();
#pragma unroll
  for (int p = 0; p < 4; ++p) {
    const int i = i0 + p * 8;
    wt[(size_t)(n0 + i) * K + k0 + j] = ts[i][j];          // coalesced over k
  }
}

// ---------------- WMMA GEMM kernels ----------------------------------------
// Block tile 128x128x32, 8 waves as 2(M) x 4(N), wave tile 64x32 (8 WMMA/step).
// A = [M][K] bf16 row-major, B = [N][K] bf16 (pre-transposed) => all staging is
// contiguous row copies => async global->LDS, double-buffered, 1 barrier/step.
#define BM 128
#define BN 128
#define BK 32
#define KP 40         // LDS row stride (bf16): 80B, keeps b128 lanes bank-spread
#define NIT (EMBED / BK)

// kernel 1: C = x@w_qkv + b, scattered to Q,K [B,H,GS,D] and V^T [B,H,D,GS]
__global__ __launch_bounds__(256)
void qkv_gemm_kernel(const __bf16* __restrict__ xb, const __bf16* __restrict__ wt,
                     const float* __restrict__ bias,
                     __bf16* __restrict__ qb, __bf16* __restrict__ kbuf,
                     __bf16* __restrict__ vtb) {
  __shared__ __bf16 As[2][BM * KP];
  __shared__ __bf16 Bs[2][BN * KP];
  const int tid   = threadIdx.x;
  const int lane  = tid & 31;
  const int wave  = tid >> 5;
  const int lrow  = lane & 15;
  const int lhalf = lane >> 4;
  const int m0 = blockIdx.y * BM;
  const int n0 = blockIdx.x * BN;
  const int waveM = (wave & 1) * 64;
  const int waveN = (wave >> 1) * 32;
  const int srow = tid >> 2;            // 0..63
  const int sc   = (tid & 3) * 8;       // 16B chunk within 32-elem row

  auto stage = [&](int k0, int buf) {
#pragma unroll
    for (int p = 0; p < 2; ++p) {
      const int row = srow + p * 64;
      copy16(xb + (size_t)(m0 + row) * EMBED + k0 + sc, &As[buf][row * KP + sc]);
      copy16(wt + (size_t)(n0 + row) * EMBED + k0 + sc, &Bs[buf][row * KP + sc]);
    }
  };

  f32x8 acc[4][2];
#pragma unroll
  for (int i = 0; i < 4; ++i)
#pragma unroll
    for (int j = 0; j < 2; ++j) acc[i][j] = zero8();

  stage(0, 0);
  for (int it = 0; it < NIT; ++it) {
    wait_stage();
    __syncthreads();                       // tile it landed; prev compute done
    if (it + 1 < NIT) stage((it + 1) * BK, (it + 1) & 1);  // overlap w/ compute
    const int buf = it & 1;
    bf16x16 af[4], bfr[2];
#pragma unroll
    for (int ms = 0; ms < 4; ++ms)
      af[ms] = loadA32(&As[buf][(waveM + ms * 16 + lrow) * KP], lhalf);
#pragma unroll
    for (int ns = 0; ns < 2; ++ns)
      bfr[ns] = loadB32(&Bs[buf][(waveN + ns * 16 + lrow) * KP], lhalf);
#pragma unroll
    for (int ms = 0; ms < 4; ++ms)
#pragma unroll
      for (int ns = 0; ns < 2; ++ns)
        acc[ms][ns] = wmma_bf16(af[ms], bfr[ns], acc[ms][ns]);
  }

#pragma unroll
  for (int ms = 0; ms < 4; ++ms)
#pragma unroll
    for (int ns = 0; ns < 2; ++ns) {
      const int n = n0 + waveN + ns * 16 + lrow;
      const float bv = bias[n];
      const int which = n / EMBED;
      const int h = (n % EMBED) / HEAD_DIM;
      const int d = n % HEAD_DIM;
#pragma unroll
      for (int r = 0; r < 8; ++r) {
        const int m = m0 + waveM + ms * 16 + lhalf * 8 + r;
        const int b = m >> 10, g = m & 1023;
        const float v = acc[ms][ns][r] + bv;
        const size_t bh = (size_t)b * HEADS + h;
        if (which == 0)      qb  [(bh * SEQ + g) * HEAD_DIM + d] = (__bf16)v;
        else if (which == 1) kbuf[(bh * SEQ + g) * HEAD_DIM + d] = (__bf16)v;
        else                 vtb [(bh * HEAD_DIM + d) * SEQ + g] = (__bf16)v;
      }
    }
}

// kernel 3: out = attn@w_proj + b (fp32 out)
__global__ __launch_bounds__(256)
void proj_gemm_kernel(const __bf16* __restrict__ a, const __bf16* __restrict__ wt,
                      const float* __restrict__ bias, float* __restrict__ out) {
  __shared__ __bf16 As[2][BM * KP];
  __shared__ __bf16 Bs[2][BN * KP];
  const int tid   = threadIdx.x;
  const int lane  = tid & 31;
  const int wave  = tid >> 5;
  const int lrow  = lane & 15;
  const int lhalf = lane >> 4;
  const int m0 = blockIdx.y * BM;
  const int n0 = blockIdx.x * BN;
  const int waveM = (wave & 1) * 64;
  const int waveN = (wave >> 1) * 32;
  const int srow = tid >> 2;
  const int sc   = (tid & 3) * 8;

  auto stage = [&](int k0, int buf) {
#pragma unroll
    for (int p = 0; p < 2; ++p) {
      const int row = srow + p * 64;
      copy16(a  + (size_t)(m0 + row) * EMBED + k0 + sc, &As[buf][row * KP + sc]);
      copy16(wt + (size_t)(n0 + row) * EMBED + k0 + sc, &Bs[buf][row * KP + sc]);
    }
  };

  f32x8 acc[4][2];
#pragma unroll
  for (int i = 0; i < 4; ++i)
#pragma unroll
    for (int j = 0; j < 2; ++j) acc[i][j] = zero8();

  stage(0, 0);
  for (int it = 0; it < NIT; ++it) {
    wait_stage();
    __syncthreads();
    if (it + 1 < NIT) stage((it + 1) * BK, (it + 1) & 1);
    const int buf = it & 1;
    bf16x16 af[4], bfr[2];
#pragma unroll
    for (int ms = 0; ms < 4; ++ms)
      af[ms] = loadA32(&As[buf][(waveM + ms * 16 + lrow) * KP], lhalf);
#pragma unroll
    for (int ns = 0; ns < 2; ++ns)
      bfr[ns] = loadB32(&Bs[buf][(waveN + ns * 16 + lrow) * KP], lhalf);
#pragma unroll
    for (int ms = 0; ms < 4; ++ms)
#pragma unroll
      for (int ns = 0; ns < 2; ++ns)
        acc[ms][ns] = wmma_bf16(af[ms], bfr[ns], acc[ms][ns]);
  }

#pragma unroll
  for (int ms = 0; ms < 4; ++ms)
#pragma unroll
    for (int ns = 0; ns < 2; ++ns) {
      const int n = n0 + waveN + ns * 16 + lrow;
      const float bv = bias[n];
#pragma unroll
      for (int r = 0; r < 8; ++r) {
        const int m = m0 + waveM + ms * 16 + lhalf * 8 + r;
        out[(size_t)m * EMBED + n] = acc[ms][ns][r] + bv;
      }
    }
}

// ---------------- kernel 2: flash attention --------------------------------
// grid (SEQ/64, B*H), 128 threads (4 waves, 16 q-rows each). K and V^T tiles
// (64x64 bf16) are cooperatively staged into double-buffered LDS (async),
// cutting VMEM traffic 4x vs per-wave fragment fetches.
#define PP  72
#define KP2 72
#define ANIT (SEQ / 64)

__global__ __launch_bounds__(128)
void attn_kernel(const __bf16* __restrict__ qb, const __bf16* __restrict__ kbuf,
                 const __bf16* __restrict__ vtb, __bf16* __restrict__ attn) {
  __shared__ __bf16 Ks[2][64 * KP2];
  __shared__ __bf16 Vs[2][64 * KP2];
  __shared__ __bf16 Pls[4 * 16 * PP];
  const int tid   = threadIdx.x;
  const int lane  = tid & 31;
  const int wave  = tid >> 5;
  const int lrow  = lane & 15;
  const int lhalf = lane >> 4;
  const int bh = blockIdx.y;
  const int q0 = blockIdx.x * 64 + wave * 16;
  const int b  = bh / HEADS, h = bh % HEADS;

  const __bf16* Q  = qb   + (size_t)bh * SEQ * HEAD_DIM;
  const __bf16* K  = kbuf + (size_t)bh * SEQ * HEAD_DIM;
  const __bf16* VT = vtb  + (size_t)bh * HEAD_DIM * SEQ;
  __bf16* Pw = &Pls[wave * 16 * PP];

  const int srow = tid >> 3;          // 0..15
  const int sc   = (tid & 7) * 8;     // 16B chunk within 64-elem row

  auto stage = [&](int kb0, int buf) {
#pragma unroll
    for (int p = 0; p < 4; ++p) {
      const int row = srow + p * 16;  // 0..63
      copy16(K  + (size_t)(kb0 + row) * HEAD_DIM + sc, &Ks[buf][row * KP2 + sc]);
      copy16(VT + (size_t)row * SEQ + kb0 + sc,        &Vs[buf][row * KP2 + sc]);
    }
  };

  bf16x16 qf[2];
#pragma unroll
  for (int c = 0; c < 2; ++c)
    qf[c] = loadA32(Q + (size_t)(q0 + lrow) * HEAD_DIM + c * 32, lhalf);

  f32x8 o[4];
#pragma unroll
  for (int ds = 0; ds < 4; ++ds) o[ds] = zero8();
  float mst[8], lst[8];
#pragma unroll
  for (int r = 0; r < 8; ++r) { mst[r] = -INFINITY; lst[r] = 0.f; }
  const float scale = 0.125f;  // HEAD_DIM^-0.5

  stage(0, 0);
  for (int it = 0; it < ANIT; ++it) {
    wait_stage();
    __syncthreads();
    if (it + 1 < ANIT) stage((it + 1) * 64, (it + 1) & 1);
    const int buf = it & 1;

    // S = Q @ K^T from LDS tiles
    f32x8 s[4];
#pragma unroll
    for (int ns = 0; ns < 4; ++ns) {
      s[ns] = zero8();
#pragma unroll
      for (int c = 0; c < 2; ++c)
        s[ns] = wmma_bf16(qf[c],
                          loadB32(&Ks[buf][(ns * 16 + lrow) * KP2 + c * 32], lhalf),
                          s[ns]);
#pragma unroll
      for (int r = 0; r < 8; ++r) s[ns][r] *= scale;
    }

    // online softmax
    float mnew[8], corr[8];
#pragma unroll
    for (int r = 0; r < 8; ++r) {
      float v = fmaxf(fmaxf(s[0][r], s[1][r]), fmaxf(s[2][r], s[3][r]));
      v = rmax16(v);
      mnew[r] = fmaxf(mst[r], v);
      corr[r] = __expf(mst[r] - mnew[r]);
      mst[r]  = mnew[r];
    }
    float psum[8];
#pragma unroll
    for (int r = 0; r < 8; ++r) psum[r] = 0.f;
#pragma unroll
    for (int ns = 0; ns < 4; ++ns)
#pragma unroll
      for (int r = 0; r < 8; ++r) {
        const float p = __expf(s[ns][r] - mnew[r]);
        psum[r] += p;
        Pw[(lhalf * 8 + r) * PP + ns * 16 + lrow] = (__bf16)p;  // C-layout -> LDS
      }
#pragma unroll
    for (int r = 0; r < 8; ++r) lst[r] = lst[r] * corr[r] + rsum16(psum[r]);
#pragma unroll
    for (int ds = 0; ds < 4; ++ds)
#pragma unroll
      for (int r = 0; r < 8; ++r) o[ds][r] *= corr[r];

    // reload P as A-fragments (wave-private LDS, DS in-order per wave)
    bf16x16 pf[2];
#pragma unroll
    for (int c = 0; c < 2; ++c)
      pf[c] = loadA32(&Pw[lrow * PP + c * 32], lhalf);

    // O += P @ V from LDS V^T tile
#pragma unroll
    for (int ds = 0; ds < 4; ++ds)
#pragma unroll
      for (int c = 0; c < 2; ++c)
        o[ds] = wmma_bf16(pf[c],
                          loadB32(&Vs[buf][(ds * 16 + lrow) * KP2 + c * 32], lhalf),
                          o[ds]);
  }

#pragma unroll
  for (int ds = 0; ds < 4; ++ds) {
    const int d = ds * 16 + lrow;
#pragma unroll
    for (int r = 0; r < 8; ++r) {
      const int qrow = q0 + lhalf * 8 + r;
      attn[((size_t)b * SEQ + qrow) * EMBED + h * HEAD_DIM + d] =
          (__bf16)(o[ds][r] / lst[r]);
    }
  }
}

// ---------------------------------------------------------------------------
extern "C" void kernel_launch(void* const* d_in, const int* in_sizes, int n_in,
                              void* d_out, int out_size, void* d_ws, size_t ws_size,
                              hipStream_t stream) {
  const float* x      = (const float*)d_in[0];
  const float* w_qkv  = (const float*)d_in[1];
  const float* b_qkv  = (const float*)d_in[2];
  const float* w_proj = (const float*)d_in[3];
  const float* b_proj = (const float*)d_in[4];
  float* out = (float*)d_out;

  __bf16* p = (__bf16*)d_ws;
  __bf16* xb     = p; p += (size_t)MTOT * EMBED;        // 12.6M
  __bf16* wqkvT  = p; p += (size_t)QKV_N * EMBED;       // [2304][768]
  __bf16* wprojT = p; p += (size_t)EMBED * EMBED;       // [768][768]
  __bf16* qb     = p; p += (size_t)BATCH * HEADS * SEQ * HEAD_DIM;
  __bf16* kb     = p; p += (size_t)BATCH * HEADS * SEQ * HEAD_DIM;
  __bf16* vtb    = p; p += (size_t)BATCH * HEADS * SEQ * HEAD_DIM;
  __bf16* attn   = p;

  cvt_kernel<<<(MTOT * EMBED) / (256 * 4), 256, 0, stream>>>(x, xb);
  transpose_cvt_kernel<<<dim3(QKV_N / 32, EMBED / 32), 256, 0, stream>>>(
      w_qkv, wqkvT, EMBED, QKV_N);
  transpose_cvt_kernel<<<dim3(EMBED / 32, EMBED / 32), 256, 0, stream>>>(
      w_proj, wprojT, EMBED, EMBED);
  qkv_gemm_kernel<<<dim3(QKV_N / BN, MTOT / BM), 256, 0, stream>>>(
      xb, wqkvT, b_qkv, qb, kb, vtb);
  attn_kernel<<<dim3(SEQ / 64, BATCH * HEADS), 128, 0, stream>>>(
      qb, kb, vtb, attn);
  proj_gemm_kernel<<<dim3(EMBED / BN, MTOT / BM), 256, 0, stream>>>(
      attn, wprojT, b_proj, out);
}
